// SelfAttention_31318901522527
// MI455X (gfx1250) — matrix operand order repaired and checked by reading, not compiled, not run on hardware
//
#include <hip/hip_runtime.h>

// ---------------------------------------------------------------------------
// Self-attention (B=4, S=2048, H=1024) for gfx1250 (MI455X), bf16 WMMA path.
//   Kernel 0: fp32 -> bf16 conversion of x and Wq/Wk/Wv (one-time, BW-bound).
//   Kernel 1: QKV projections, ping-pong double-buffered bf16 WMMA GEMM
//             (manual unroll-by-2 + peeled final step: no clamp, no moves).
//             Q/K row-major bf16; V stored transposed [B][H][S] bf16.
//   Kernel 2: flash attention per 16-query tile: S^T = K Q^T, out^T = V^T P^T
//             so softmax stats/rescale are per-lane uniform (one shfl_xor(16)).
// ---------------------------------------------------------------------------

typedef __attribute__((ext_vector_type(16))) __bf16 v16bf;
typedef __attribute__((ext_vector_type(8)))  float  v8f;

#define DEVINL __device__ __forceinline__

constexpr int Bb = 4;
constexpr int S  = 2048;
constexpr int H  = 1024;
constexpr int BS = Bb * S;   // 8192 flattened rows

union BF16x16 { v16bf v; __bf16 e[16]; uint4 q[2]; };
union BF16x8  { __bf16 e[8]; uint4 q; };

DEVINL v8f wmma_bf16(v16bf a, v16bf b, v8f c) {
  // emits v_wmma_f32_16x16x32_bf16
  return __builtin_amdgcn_wmma_f32_16x16x32_bf16(false, a, false, b, (short)0, c,
                                                 false, false);
}

DEVINL v8f zero8() { v8f z = {0.f,0.f,0.f,0.f,0.f,0.f,0.f,0.f}; return z; }

// A-fragment (16x32 bf16, M x K). Lane = (h = lane/16, m = lane%16).
// Element i -> K = 16*h + i. `p` = row m at kbase (kbase%32==0),
// `off` a (preferably constant) extra element offset.
DEVINL v16bf load_a_bf16(const __bf16* p, size_t off, int h) {
  BF16x16 u;
  const uint4* q = reinterpret_cast<const uint4*>(p + off);
  u.q[0] = q[2 * h];
  u.q[1] = q[2 * h + 1];
  return u.v;
}

// B-fragment (32x16 bf16, K x N). Lane = (kh = lane/16, n = lane%16).
// Element i -> K = 16*kh + i. `p + off` = column n's data at K = kbase + 16*kh.
DEVINL v16bf load_b_bf16(const __bf16* p, size_t off) {
  BF16x16 u;
  const uint4* q = reinterpret_cast<const uint4*>(p + off);
  u.q[0] = q[0];
  u.q[1] = q[1];
  return u.v;
}

// ---------------------------------------------------------------------------
// Kernel 0: fp32 -> bf16, 8 elements/thread, packed 16B stores.
// ---------------------------------------------------------------------------
__global__ void __launch_bounds__(256)
cvt_bf16_kernel(const float* __restrict__ src, __bf16* __restrict__ dst, int n8)
{
  const int i = blockIdx.x * blockDim.x + threadIdx.x;
  if (i >= n8) return;
  const float4* s = reinterpret_cast<const float4*>(src);
  float4 a = s[2 * i], b = s[2 * i + 1];
  BF16x8 u;
  u.e[0] = (__bf16)a.x; u.e[1] = (__bf16)a.y; u.e[2] = (__bf16)a.z; u.e[3] = (__bf16)a.w;
  u.e[4] = (__bf16)b.x; u.e[5] = (__bf16)b.y; u.e[6] = (__bf16)b.z; u.e[7] = (__bf16)b.w;
  reinterpret_cast<uint4*>(dst)[i] = u.q;
}

// ---------------------------------------------------------------------------
// Kernel 1: QKV projection. grid = (BS/16, H/512, 3); block = 256 (8 waves).
// Wave w owns rows [16*bx,+16) x cols [512*by + 64*w, +64) of Q/K/V (bz).
// z<2 (Q,K):  C = W x^T  (M=col, N=row)  -> packed row-major stores.
// z==2 (V):   C = x W^T  (M=row, N=col)  -> packed transposed [B][H][S] stores.
// ---------------------------------------------------------------------------
__global__ void __launch_bounds__(256)
qkv_kernel(const __bf16* __restrict__ xb,
           const __bf16* __restrict__ Wqb, const float* __restrict__ bq,
           const __bf16* __restrict__ Wkb, const float* __restrict__ bk,
           const __bf16* __restrict__ Wvb, const float* __restrict__ bv,
           __bf16* __restrict__ Qo, __bf16* __restrict__ Ko,
           __bf16* __restrict__ Vt)
{
  const int tid  = threadIdx.x;
  const int lane = tid & 31;
  const int w    = tid >> 5;
  const int n    = lane & 15;
  const int hs   = lane >> 4;

  const int g0 = blockIdx.x * 16;            // flattened (b*S+s) row base
  const int n0 = blockIdx.y * 512 + w * 64;  // output column base
  const int z  = blockIdx.z;                 // 0=Q 1=K 2=V

  v8f acc[4];
#pragma unroll
  for (int s2 = 0; s2 < 4; ++s2) acc[s2] = zero8();

  if (z != 2) {
    // ---- Q/K: A = W rows (cols of output), B = x^T ----
    const __bf16* Wb   = z ? Wkb : Wqb;
    const float*  bias = z ? bk : bq;
    __bf16*       Out  = z ? Ko : Qo;

    const __bf16* xrow  = xb + (size_t)(g0 + n) * H + 16 * hs;  // B source
    const __bf16* wbase = Wb + (size_t)(n0 + n) * H;            // A source

    v16bf b0, b1, a0[4], a1[4];
    // Preload step 0 into buffer 0 (B first, A after).
    b0 = load_b_bf16(xrow, 0);
#pragma unroll
    for (int s2 = 0; s2 < 4; ++s2)
      a0[s2] = load_a_bf16(wbase, (size_t)(16 * s2) * H, hs);

    // Steady state: 15 double-steps, loads always one step ahead.
    for (int kk = 0; kk < H - 64; kk += 64) {
      b1 = load_b_bf16(xrow, kk + 32);
#pragma unroll
      for (int s2 = 0; s2 < 4; ++s2)
        a1[s2] = load_a_bf16(wbase, (size_t)(16 * s2) * H + kk + 32, hs);
#pragma unroll
      for (int s2 = 0; s2 < 4; ++s2)
        acc[s2] = wmma_bf16(a0[s2], b0, acc[s2]);

      b0 = load_b_bf16(xrow, kk + 64);
#pragma unroll
      for (int s2 = 0; s2 < 4; ++s2)
        a0[s2] = load_a_bf16(wbase, (size_t)(16 * s2) * H + kk + 64, hs);
#pragma unroll
      for (int s2 = 0; s2 < 4; ++s2)
        acc[s2] = wmma_bf16(a1[s2], b1, acc[s2]);
    }
    // Peeled final double-step (no trailing loads, no clamp).
    b1 = load_b_bf16(xrow, H - 32);
#pragma unroll
    for (int s2 = 0; s2 < 4; ++s2)
      a1[s2] = load_a_bf16(wbase, (size_t)(16 * s2) * H + H - 32, hs);
#pragma unroll
    for (int s2 = 0; s2 < 4; ++s2)
      acc[s2] = wmma_bf16(a0[s2], b0, acc[s2]);
#pragma unroll
    for (int s2 = 0; s2 < 4; ++s2)
      acc[s2] = wmma_bf16(a1[s2], b1, acc[s2]);

    // Epilogue: lane holds row n, cols colb..colb+7 -> one 16B store per tile.
    const int row = g0 + n;
#pragma unroll
    for (int s2 = 0; s2 < 4; ++s2) {
      const int colb = n0 + 16 * s2 + 8 * hs;
      float4 c0 = *reinterpret_cast<const float4*>(bias + colb);
      float4 c1 = *reinterpret_cast<const float4*>(bias + colb + 4);
      const float bvv[8] = {c0.x, c0.y, c0.z, c0.w, c1.x, c1.y, c1.z, c1.w};
      BF16x8 u;
#pragma unroll
      for (int r = 0; r < 8; ++r) u.e[r] = (__bf16)(acc[s2][r] + bvv[r]);
      *reinterpret_cast<uint4*>(Out + (size_t)row * H + colb) = u.q;
    }
  } else {
    // ---- V: A = x rows, B = W^T; store transposed [B][H][S] ----
    const __bf16* xrow  = xb + (size_t)(g0 + n) * H;                 // A source
    const __bf16* wbase = Wvb + (size_t)(n0 + n) * H + 16 * hs;      // B source

    v16bf a0, a1, b0[4], b1[4];
    a0 = load_a_bf16(xrow, 0, hs);
#pragma unroll
    for (int s2 = 0; s2 < 4; ++s2)
      b0[s2] = load_b_bf16(wbase, (size_t)(16 * s2) * H);

    for (int kk = 0; kk < H - 64; kk += 64) {
      a1 = load_a_bf16(xrow, kk + 32, hs);
#pragma unroll
      for (int s2 = 0; s2 < 4; ++s2)
        b1[s2] = load_b_bf16(wbase, (size_t)(16 * s2) * H + kk + 32);
#pragma unroll
      for (int s2 = 0; s2 < 4; ++s2)
        acc[s2] = wmma_bf16(a0, b0[s2], acc[s2]);

      a0 = load_a_bf16(xrow, kk + 64, hs);
#pragma unroll
      for (int s2 = 0; s2 < 4; ++s2)
        b0[s2] = load_b_bf16(wbase, (size_t)(16 * s2) * H + kk + 64);
#pragma unroll
      for (int s2 = 0; s2 < 4; ++s2)
        acc[s2] = wmma_bf16(a1, b1[s2], acc[s2]);
    }
    a1 = load_a_bf16(xrow, H - 32, hs);
#pragma unroll
    for (int s2 = 0; s2 < 4; ++s2)
      b1[s2] = load_b_bf16(wbase, (size_t)(16 * s2) * H + H - 32);
#pragma unroll
    for (int s2 = 0; s2 < 4; ++s2)
      acc[s2] = wmma_bf16(a0, b0[s2], acc[s2]);
#pragma unroll
    for (int s2 = 0; s2 < 4; ++s2)
      acc[s2] = wmma_bf16(a1, b1[s2], acc[s2]);

    // Epilogue: lane holds col `colc`, rows s..s+7 -> one 16B store per tile.
    const int bb = g0 / S, ss0 = g0 % S;
#pragma unroll
    for (int s2 = 0; s2 < 4; ++s2) {
      const int colc = n0 + 16 * s2 + n;
      const float bval = bv[colc];
      BF16x8 u;
#pragma unroll
      for (int r = 0; r < 8; ++r) u.e[r] = (__bf16)(acc[s2][r] + bval);
      *reinterpret_cast<uint4*>(Vt + ((size_t)bb * H + colc) * S + ss0 + 8 * hs) = u.q;
    }
  }
}

// ---------------------------------------------------------------------------
// Kernel 2: flash attention. grid = BS/16 blocks of 256 threads (8 waves).
// Block owns 16 queries; wave w owns d-chunk [128w, 128w+128).
// ---------------------------------------------------------------------------
__global__ void __launch_bounds__(256)
attn_kernel(const __bf16* __restrict__ Q, const __bf16* __restrict__ K,
            const __bf16* __restrict__ Vt, float* __restrict__ out)
{
  // Partial-score buffer: [keytile t][m=key][n=query][wave] -> vector reads.
  __shared__ float pbuf[2 * 16 * 16 * 8];

  const int tid  = threadIdx.x;
  const int lane = tid & 31;
  const int w    = tid >> 5;
  const int n    = lane & 15;
  const int hs   = lane >> 4;

  const int grow0 = blockIdx.x * 16;     // flattened (b*S+s) query base
  const int b     = grow0 / S;

  // Q^T B-fragments for this wave's 128-wide d-chunk stay in registers.
  v16bf qb[4];
  {
    const __bf16* qrow = Q + (size_t)(grow0 + n) * H + 128 * w + 16 * hs;
#pragma unroll
    for (int ks = 0; ks < 4; ++ks)
      qb[ks] = load_b_bf16(qrow, 32 * ks);
  }

  v8f oacc[8];
#pragma unroll
  for (int s2 = 0; s2 < 8; ++s2) oacc[s2] = zero8();
  float m_run = -3.0e38f;
  float ssum  = 0.f;

  const __bf16* Kb = K  + (size_t)b * S * H + 128 * w;
  const __bf16* Vb = Vt + (size_t)b * H * S + (size_t)(128 * w) * S;

  for (int j0 = 0; j0 < S; j0 += 32) {
    // Warm caches for the next key tile while this one computes.
    if (j0 + 32 < S) {
      __builtin_prefetch(Kb + (size_t)(j0 + 32 + n) * H, 0, 0);
      __builtin_prefetch(Kb + (size_t)(j0 + 48 + n) * H, 0, 0);
      __builtin_prefetch(Vb + (size_t)n * S + j0 + 32, 0, 0);
      __builtin_prefetch(Vb + (size_t)(64 + n) * S + j0 + 32, 0, 0);
    }

    // --- partial scores over this wave's d-chunk (keys j0..j0+31) ---
    v16bf ka[8];
    {
      const __bf16* k0 = Kb + (size_t)(j0 + n) * H;
#pragma unroll
      for (int ks = 0; ks < 4; ++ks) {
        ka[2 * ks + 0] = load_a_bf16(k0, 32 * ks, hs);
        ka[2 * ks + 1] = load_a_bf16(k0, (size_t)16 * H + 32 * ks, hs);
      }
    }
    v8f s0 = zero8(), s1 = zero8();
#pragma unroll
    for (int ks = 0; ks < 4; ++ks) {
      s0 = wmma_bf16(ka[2 * ks + 0], qb[ks], s0);
      s1 = wmma_bf16(ka[2 * ks + 1], qb[ks], s1);
    }

    // Issue V^T fragment loads now; latency hides behind LDS reduction,
    // barriers and softmax transcendentals.
    v16bf va[8];
    {
      const __bf16* v0 = Vb + (size_t)n * S + j0;
#pragma unroll
      for (int s2 = 0; s2 < 8; ++s2)
        va[s2] = load_a_bf16(v0, (size_t)(16 * s2) * S, hs);
    }

    // --- cross-wave reduction of the H-split partials through LDS ---
#pragma unroll
    for (int r = 0; r < 8; ++r) {
      const int m = r + 8 * hs;
      pbuf[((0 * 16 + m) * 16 + n) * 8 + w] = s0[r];
      pbuf[((1 * 16 + m) * 16 + n) * 8 + w] = s1[r];
    }
    __syncthreads();
    float st0[8], st1[8];
#pragma unroll
    for (int r = 0; r < 8; ++r) {
      const int m = r + 8 * hs;
      const float4* p0 = reinterpret_cast<const float4*>(&pbuf[((0 * 16 + m) * 16 + n) * 8]);
      const float4* p1 = reinterpret_cast<const float4*>(&pbuf[((1 * 16 + m) * 16 + n) * 8]);
      float4 a0 = p0[0], a1 = p0[1], c0 = p1[0], c1 = p1[1];
      st0[r] = ((a0.x + a0.y) + (a0.z + a0.w)) + ((a1.x + a1.y) + (a1.z + a1.w));
      st1[r] = ((c0.x + c0.y) + (c0.z + c0.w)) + ((c1.x + c1.y) + (c1.z + c1.w));
    }
    __syncthreads();   // pbuf reusable next iteration

    // --- online softmax (identical redundant math in every wave) ---
    float tmax = st0[0];
#pragma unroll
    for (int r = 0; r < 8; ++r) {
      tmax = fmaxf(tmax, st0[r]);
      tmax = fmaxf(tmax, st1[r]);
    }
    tmax = fmaxf(tmax, __shfl_xor(tmax, 16, 32));
    const float mnew  = fmaxf(m_run, tmax);
    const float alpha = __expf(m_run - mnew);
    float pt0[8], pt1[8], psum = 0.f;
#pragma unroll
    for (int r = 0; r < 8; ++r) {
      pt0[r] = __expf(st0[r] - mnew);
      pt1[r] = __expf(st1[r] - mnew);
      psum += pt0[r] + pt1[r];
    }
    psum += __shfl_xor(psum, 16, 32);
    ssum  = ssum * alpha + psum;
    m_run = mnew;

    // --- build P^T B-fragment (32 keys x 16 queries) via half-swap shfl ---
    BF16x16 pb;
#pragma unroll
    for (int r = 0; r < 8; ++r) {
      const float o0 = __shfl_xor(pt0[r], 16, 32);
      const float o1 = __shfl_xor(pt1[r], 16, 32);
      pb.e[r]     = (__bf16)(hs ? o1 : pt0[r]);   // K = 16*kh + r
      pb.e[8 + r] = (__bf16)(hs ? pt1[r] : o0);   // K = 16*kh + 8 + r
    }

    // --- rescale and accumulate out^T += V^T x P^T for 8 d-subtiles ---
#pragma unroll
    for (int s2 = 0; s2 < 8; ++s2) {
#pragma unroll
      for (int r = 0; r < 8; ++r) oacc[s2][r] *= alpha;
      oacc[s2] = wmma_bf16(va[s2], pb.v, oacc[s2]);
    }
  }

  // --- normalize and store fp32 output ---
  const float inv = 1.0f / ssum;
#pragma unroll
  for (int s2 = 0; s2 < 8; ++s2) {
#pragma unroll
    for (int r = 0; r < 8; ++r) {
      const int d = 128 * w + 16 * s2 + r + 8 * hs;
      out[((size_t)grow0 + n) * H + d] = oacc[s2][r] * inv;
    }
  }
}

// ---------------------------------------------------------------------------
extern "C" void kernel_launch(void* const* d_in, const int* in_sizes, int n_in,
                              void* d_out, int out_size, void* d_ws,
                              size_t ws_size, hipStream_t stream) {
  (void)in_sizes; (void)n_in; (void)out_size; (void)ws_size;

  const float* x  = (const float*)d_in[0];
  const float* Wq = (const float*)d_in[1];
  const float* bq = (const float*)d_in[2];
  const float* Wk = (const float*)d_in[3];
  const float* bk = (const float*)d_in[4];
  const float* Wv = (const float*)d_in[5];
  const float* bv = (const float*)d_in[6];
  float* out = (float*)d_out;

  // Workspace (bf16): x (16MB) | Wq|Wk|Wv (2MB each) | Q | K | V^T (16MB each)
  __bf16* xb  = (__bf16*)d_ws;
  __bf16* Wqb = xb  + (size_t)BS * H;
  __bf16* Wkb = Wqb + (size_t)H * H;
  __bf16* Wvb = Wkb + (size_t)H * H;
  __bf16* Qw  = Wvb + (size_t)H * H;
  __bf16* Kw  = Qw  + (size_t)BS * H;
  __bf16* Vw  = Kw  + (size_t)BS * H;

  const int nx8 = (BS * H) / 8, nw8 = (H * H) / 8;
  cvt_bf16_kernel<<<dim3((nx8 + 255) / 256), 256, 0, stream>>>(x, xb, nx8);
  cvt_bf16_kernel<<<dim3((nw8 + 255) / 256), 256, 0, stream>>>(Wq, Wqb, nw8);
  cvt_bf16_kernel<<<dim3((nw8 + 255) / 256), 256, 0, stream>>>(Wk, Wkb, nw8);
  cvt_bf16_kernel<<<dim3((nw8 + 255) / 256), 256, 0, stream>>>(Wv, Wvb, nw8);

  qkv_kernel<<<dim3(BS / 16, H / 512, 3), 256, 0, stream>>>(
      xb, Wqb, bq, Wkb, bk, Wvb, bv, Qw, Kw, Vw);
  attn_kernel<<<dim3(BS / 16), 256, 0, stream>>>(Qw, Kw, Vw, out);
}
